// KNearestPatchesEmbedding_56289841381640
// MI455X (gfx1250) — matrix operand order repaired and checked by reading, not compiled, not run on hardware
//
#include <hip/hip_runtime.h>

typedef __attribute__((ext_vector_type(16))) _Float16 v16h;
typedef __attribute__((ext_vector_type(8)))  float    v8f;

#define NIMG 32
#define CIN 3
#define HW 64
#define NCH 1024
#define KDIM 75
#define K96 96
#define PADW 2
#define TPIX 16
#define THREADS 128
#define NWAVE 4

union U16H {
  v16h v;
  _Float16 h[16];
  unsigned u[8];
  uint4 q[2];
};

struct BFrag {
  U16H b[3];  // 3 k-steps of 32
};

__device__ inline unsigned fkey(float f) {
  unsigned u = __float_as_uint(f);
  return (u & 0x80000000u) ? ~u : (u | 0x80000000u);
}

__device__ inline void loadB(BFrag& f, const _Float16* __restrict__ wB,
                             int ch, int hi) {
#pragma unroll
  for (int ks = 0; ks < 3; ++ks) {
    // lane holds column n=ch, 16 contiguous K halves at K = ks*32 + 16*hi
    const uint4* bp = (const uint4*)(wB + ch * K96 + ks * 32 + hi * 16);
    f.b[ks].q[0] = bp[0];
    f.b[ks].q[1] = bp[1];
  }
}

// Convert (1024 x 75) f32 conv weights to f16, K zero-padded to 96.
__global__ void prep_weights_f16(const float* __restrict__ w,
                                 _Float16* __restrict__ wB) {
  int idx = blockIdx.x * blockDim.x + threadIdx.x;
  if (idx >= NCH * K96) return;
  int c = idx / K96, k = idx - c * K96;
  float v = (k < KDIM) ? w[c * KDIM + k] : 0.f;
  wB[idx] = (_Float16)v;
}

__global__ __launch_bounds__(THREADS) void knn_mask_kernel(
    const float* __restrict__ images,
    const _Float16* __restrict__ wB,
    const float* __restrict__ bias,
    const int* __restrict__ kptr,
    float* __restrict__ out) {
  extern __shared__ char smem[];
  float* dist = (float*)smem;                                      // 16 x 1024 f32 (64 KB)
  unsigned* kthU = (unsigned*)(smem + TPIX * NCH * sizeof(float)); // 16 u32
  _Float16* aT = (_Float16*)smem;  // alias: 16 x 96 f16 im2col tile, dead after frag load

  const int tid = threadIdx.x;
  const int wv = tid >> 5;
  const int lane = tid & 31;
  const int hi = lane >> 4;
  const int l15 = lane & 15;

  // tile coords: 16 consecutive w in one row of one image
  const int tile = blockIdx.x;
  const int w0 = (tile & 3) * TPIX;
  const int h = (tile >> 2) & (HW - 1);
  const int n = tile >> 8;

  // Preload this wave's 16 bias values (one clause, overlaps im2col below)
  float bv[16];
#pragma unroll
  for (int i = 0; i < 16; ++i) bv[i] = bias[(wv + NWAVE * i) * 16 + l15];

  // 1) im2col 16 pixels x K96 into LDS (f16), zero pad OOB and K >= 75
  for (int idx = tid; idx < TPIX * K96; idx += THREADS) {
    int p = idx / K96, k = idx - p * K96;
    float v = 0.f;
    if (k < KDIM) {
      int ci = k / 25, rem = k - ci * 25, r = rem / 5, s = rem - r * 5;
      int hh = h - PADW + r, ww = w0 + p - PADW + s;
      if (hh >= 0 && hh < HW && ww >= 0 && ww < HW)
        v = images[((n * CIN + ci) * HW + hh) * HW + ww];
    }
    aT[idx] = (_Float16)v;
  }
  __syncthreads();

  // 2) A fragments (3 k-steps), contiguous K order matching B:
  //    lanes 0-15: M=lane, K {0..15};
  //    lanes 16-31: same M, K {16..31}.
  U16H afrag[3];
  const unsigned* aU = (const unsigned*)aT;
#pragma unroll
  for (int ks = 0; ks < 3; ++ks) {
#pragma unroll
    for (int v = 0; v < 8; ++v) {
      int k0 = 16 * hi + 2 * v;
      afrag[ks].u[v] = aU[(l15 * K96 + ks * 32 + k0) >> 1];
    }
  }
  __syncthreads();  // aT now dead; dist region writable

  // 3) GEMM: each wave does 16 channel tiles (3 WMMAs each, K=96),
  //    with register double-buffering of B so loads for tile i+1 are in
  //    flight during the WMMAs of tile i.
  BFrag B0, B1;
  loadB(B0, wB, (wv + 0) * 16 + l15, hi);
#pragma unroll
  for (int ii = 0; ii < 8; ++ii) {
    const int i0 = 2 * ii, i1 = 2 * ii + 1;
    loadB(B1, wB, (wv + NWAVE * i1) * 16 + l15, hi);
    {
      v8f acc = {};
#pragma unroll
      for (int ks = 0; ks < 3; ++ks)
        acc = __builtin_amdgcn_wmma_f32_16x16x32_f16(false, afrag[ks].v, false,
                                                     B0.b[ks].v, (short)0, acc,
                                                     false, false);
      const int ch = (wv + NWAVE * i0) * 16 + l15;
#pragma unroll
      for (int r = 0; r < 8; ++r)
        dist[(r + 8 * hi) * NCH + ch] = acc[r] + bv[i0];
    }
    if (ii < 7) loadB(B0, wB, (wv + NWAVE * (i1 + 1)) * 16 + l15, hi);
    {
      v8f acc = {};
#pragma unroll
      for (int ks = 0; ks < 3; ++ks)
        acc = __builtin_amdgcn_wmma_f32_16x16x32_f16(false, afrag[ks].v, false,
                                                     B1.b[ks].v, (short)0, acc,
                                                     false, false);
      const int ch = (wv + NWAVE * i1) * 16 + l15;
#pragma unroll
      for (int r = 0; r < 8; ++r)
        dist[(r + 8 * hi) * NCH + ch] = acc[r] + bv[i1];
    }
  }
  __syncthreads();

  // 4) exact k-th smallest per pixel via binary search over order-preserving u32
  const int kk = *kptr;
  for (int q = 0; q < 4; ++q) {
    int p = wv * 4 + q;
    unsigned vu[32];
#pragma unroll
    for (int j = 0; j < 32; ++j)
      vu[j] = fkey(dist[p * NCH + j * 32 + lane]);  // lane-contiguous: no bank conflicts
    unsigned lo = 0u, hiB = 0xFFFFFFFFu;
    for (int it = 0; it < 32; ++it) {
      unsigned mid = lo + ((hiB - lo) >> 1);
      int cnt = 0;
#pragma unroll
      for (int j = 0; j < 32; ++j) cnt += (vu[j] <= mid) ? 1 : 0;
#pragma unroll
      for (int off = 16; off >= 1; off >>= 1)
        cnt += __shfl_xor(cnt, off, 32);  // wave32 reduction
      if (cnt >= kk) hiB = mid; else lo = mid + 1u;
    }
    if (lane == 0) kthU[p] = lo;
  }
  __syncthreads();

  // 5) mask write: the only HBM-heavy stream (512 MB total), b128 stores
  unsigned kq[16];
#pragma unroll
  for (int p = 0; p < 16; ++p) kq[p] = kthU[p];
  const long long obase = ((long long)n * NCH) * (HW * HW) + h * HW + w0;
  for (int c = tid; c < NCH; c += THREADS) {
    float m[16];
#pragma unroll
    for (int p = 0; p < 16; ++p)
      m[p] = (fkey(dist[p * NCH + c]) <= kq[p]) ? 1.0f : 0.0f;
    float4* o4 = (float4*)(out + obase + (long long)c * (HW * HW));
    o4[0] = make_float4(m[0], m[1], m[2], m[3]);
    o4[1] = make_float4(m[4], m[5], m[6], m[7]);
    o4[2] = make_float4(m[8], m[9], m[10], m[11]);
    o4[3] = make_float4(m[12], m[13], m[14], m[15]);
  }
}

extern "C" void kernel_launch(void* const* d_in, const int* in_sizes, int n_in,
                              void* d_out, int out_size, void* d_ws, size_t ws_size,
                              hipStream_t stream) {
  const float* images = (const float*)d_in[0];
  const float* kern = (const float*)d_in[1];
  const float* bias = (const float*)d_in[2];
  const int* kptr = (const int*)d_in[3];
  float* out = (float*)d_out;
  _Float16* wB = (_Float16*)d_ws;  // 1024*96*2 = 192 KB of workspace

  prep_weights_f16<<<(NCH * K96 + 255) / 256, 256, 0, stream>>>(kern, wB);

  const size_t smemBytes = TPIX * NCH * sizeof(float) + 64;
  const int ntiles = NIMG * HW * (HW / TPIX);  // 8192 workgroups
  knn_mask_kernel<<<ntiles, THREADS, smemBytes, stream>>>(images, wB, bias, kptr, out);
}
